// PositionwiseSelfAttention_31756988187109
// MI455X (gfx1250) — matrix operand order repaired and checked by reading, not compiled, not run on hardware
//
#include <hip/hip_runtime.h>
#include <hip/hip_bf16.h>

// ---------------------------------------------------------------------------
// PositionwiseSelfAttention on MI455X (gfx1250, wave32, WMMA).
// B=32, S=4096, E=256, H=8 (dh=32), P=256, M=16, G=B*P=8192 groups.
// Memory-bound (~258MB HBM ~= 11us floor @ 23.3TB/s): all matmuls on
// v_wmma_f32_16x16x32_f16 (f32 accum), one wave32 per group, LDS-staged.
// A-fragments for the 16x256 activation matrix are hoisted into 64 VGPRs
// and reused across the Q, K and V GEMMs.
// ---------------------------------------------------------------------------

typedef _Float16 half16_t __attribute__((ext_vector_type(16)));
typedef _Float16 half8_t  __attribute__((ext_vector_type(8)));
typedef float    float8_t __attribute__((ext_vector_type(8)));

#define BB 32
#define SS 4096
#define EE 256
#define HH 8
#define DH 32
#define PP 256
#define MM 16
#define GG (BB * PP)
#define XPAD 272   // 16-half pad on 256-half rows: keeps 32B alignment, breaks bank stride

__device__ __forceinline__ half16_t cat8(half8_t a, half8_t b) {
  return __builtin_shufflevector(a, b, 0,1,2,3,4,5,6,7,8,9,10,11,12,13,14,15);
}

// 16-bit A-matrix fragment (16x32, M=lane&15) from an LDS row pointer at k-base.
// Contiguous layout: lane<16 holds K = 0..15, lane>=16 holds K = 16..31.
__device__ __forceinline__ half16_t load_a_frag(const _Float16* row_kb, int hi) {
  const _Float16* p = row_kb + hi * 16;
  half8_t lo  = *(const half8_t*)p;        // K = kb + hi*16 + 0..7
  half8_t hi8 = *(const half8_t*)(p + 8);  // K = kb + hi*16 + 8..15
  return cat8(lo, hi8);
}

// ---------------------------------------------------------------------------
// Kernel 1: pack fp32 weights -> f16 B-fragment-major layout.
// dst[tid], tid bits: e=tid[3:0], L=tid[8:4], s=tid[11:9], t=tid[15:12], w=tid[16+]
// B-fragment element e of lane L (tile t, k-step s): k = s*32 + (L>>4)*16 + e,
// n = t*16 + (L&15). Consumer reads 16 contiguous halfs (32B) per lane.
// ---------------------------------------------------------------------------
__global__ void pack_weights(const float* __restrict__ wq, const float* __restrict__ wk,
                             const float* __restrict__ wv, const float* __restrict__ w0,
                             _Float16* __restrict__ dst) {
  int tid = blockIdx.x * blockDim.x + threadIdx.x;
  if (tid >= 4 * 65536) return;
  int e = tid & 15;
  int L = (tid >> 4) & 31;
  int s = (tid >> 9) & 7;
  int t = (tid >> 12) & 15;
  int w = tid >> 16;
  int k = s * 32 + (L >> 4) * 16 + e;
  int n = t * 16 + (L & 15);
  const float* W = (w == 0) ? wq : (w == 1) ? wk : (w == 2) ? wv : w0;
  dst[tid] = (_Float16)W[k * EE + n];
}

// ---------------------------------------------------------------------------
// Kernel 2: build per-group token lists (stable order == reference rank).
// One block per batch row; thread p scans the row once, recording up to M
// token indices whose position == p, in original order.
// ---------------------------------------------------------------------------
__global__ void build_groups(const int* __restrict__ positions, int* __restrict__ sidx) {
  __shared__ int row[SS];
  const int b = blockIdx.x;
  for (int i = threadIdx.x; i < SS; i += blockDim.x) row[i] = positions[b * SS + i];
  __syncthreads();
  const int p = threadIdx.x;           // position 0..255
  const int g = b * PP + p;
  int cnt = 0;
  for (int i = 0; i < SS; ++i) {
    if (row[i] == p) {
      if (cnt < MM) sidx[g * MM + cnt] = i;
      ++cnt;
    }
  }
  for (int m = (cnt < MM ? cnt : MM); m < MM; ++m) sidx[g * MM + m] = -1;
}

// ---------------------------------------------------------------------------
// Kernel 3: one wave32 per group.
// ---------------------------------------------------------------------------
__global__ __launch_bounds__(32) void pw_attn(const float* __restrict__ emb,
                                              const _Float16* __restrict__ pw,
                                              const int* __restrict__ sidx_g,
                                              float* __restrict__ out) {
  __shared__ __align__(32) _Float16 Xs[MM][XPAD];   // X, later reused as ctx
  __shared__ __align__(32) _Float16 Qs[MM][XPAD];
  __shared__ __align__(32) _Float16 Ks[MM][XPAD];
  __shared__ __align__(32) _Float16 Vt[EE][16];     // V transposed: [d][token]
  __shared__ float sc[16][16];
  __shared__ float rmax[16], rrcp[16];
  __shared__ __align__(32) _Float16 attnp[16][40];  // K-padded attn probs (cols 16..31 = 0)
  __shared__ int sidx[MM];

  const int g    = blockIdx.x;
  const int lane = threadIdx.x & 31;
  const int ln   = lane & 15;
  const int hi   = lane >> 4;
  const int b    = g >> 8;               // g / P

  if (lane < MM) sidx[lane] = sidx_g[g * MM + lane];
  __syncthreads();

  // ---- gather 16 token rows, fp32 -> f16, coalesced ----
  for (int m = 0; m < MM; ++m) {
    const int tok = sidx[m];
    const float* src = emb + ((size_t)(b * SS + (tok < 0 ? 0 : tok))) * EE;
    for (int c = lane; c < EE; c += 32) {
      float v = 0.0f;
      if (tok >= 0) v = src[c];
      Xs[m][c] = (_Float16)v;
    }
  }
  __syncthreads();

  // ---- hoist all 8 A-fragments of X into registers (64 VGPRs), reuse for Q/K/V ----
  half16_t Af[8];
#pragma unroll
  for (int s = 0; s < 8; ++s) Af[s] = load_a_frag(&Xs[ln][s * 32], hi);

  // ---- Q projection ----
#pragma unroll 1
  for (int t = 0; t < 16; ++t) {
    float8_t acc = {};
#pragma unroll
    for (int s = 0; s < 8; ++s) {
      const _Float16* bp = pw + ((((size_t)0 * 16 + t) * 8 + s) * 32 + lane) * 16;
      half16_t Bv = *(const half16_t*)bp;
      acc = __builtin_amdgcn_wmma_f32_16x16x32_f16(false, Af[s], false, Bv,
                                                   (short)0, acc, false, false);
    }
#pragma unroll
    for (int r = 0; r < 8; ++r) Qs[r + 8 * hi][t * 16 + ln] = (_Float16)acc[r];
  }

  // ---- K projection ----
#pragma unroll 1
  for (int t = 0; t < 16; ++t) {
    float8_t acc = {};
#pragma unroll
    for (int s = 0; s < 8; ++s) {
      const _Float16* bp = pw + ((((size_t)1 * 16 + t) * 8 + s) * 32 + lane) * 16;
      half16_t Bv = *(const half16_t*)bp;
      acc = __builtin_amdgcn_wmma_f32_16x16x32_f16(false, Af[s], false, Bv,
                                                   (short)0, acc, false, false);
    }
#pragma unroll
    for (int r = 0; r < 8; ++r) Ks[r + 8 * hi][t * 16 + ln] = (_Float16)acc[r];
  }

  // ---- V projection (stored transposed: Vt[d][token]) ----
#pragma unroll 1
  for (int t = 0; t < 16; ++t) {
    float8_t acc = {};
#pragma unroll
    for (int s = 0; s < 8; ++s) {
      const _Float16* bp = pw + ((((size_t)2 * 16 + t) * 8 + s) * 32 + lane) * 16;
      half16_t Bv = *(const half16_t*)bp;
      acc = __builtin_amdgcn_wmma_f32_16x16x32_f16(false, Af[s], false, Bv,
                                                   (short)0, acc, false, false);
    }
#pragma unroll
    for (int r = 0; r < 8; ++r) Vt[t * 16 + ln][r + 8 * hi] = (_Float16)acc[r];
  }
  __syncthreads();

  // ---- per-head attention: scores (1 wmma), f32 softmax, ctx (2 wmma) ----
#pragma unroll 1
  for (int h = 0; h < HH; ++h) {
    float8_t acc = {};
    {
      half16_t A = load_a_frag(&Qs[ln][h * DH], hi);           // Q_h 16x32
      // B = K_h^T (32x16): lane = token n, element e -> dh index e + hi*16 (contiguous)
      half16_t Bv = *(const half16_t*)&Ks[ln][h * DH + hi * 16];
      acc = __builtin_amdgcn_wmma_f32_16x16x32_f16(false, A, false, Bv,
                                                   (short)0, acc, false, false);
    }
    const float scale = 0.17677669529663687f;                  // 1/sqrt(32)
#pragma unroll
    for (int r = 0; r < 8; ++r) sc[r + 8 * hi][ln] = acc[r] * scale;
    __syncthreads();
    if (lane < 16) {
      float mx = sc[lane][0];
      for (int j = 1; j < 16; ++j) mx = fmaxf(mx, sc[lane][j]);
      float sum = 0.0f;
      for (int j = 0; j < 16; ++j) sum += __expf(sc[lane][j] - mx);
      rmax[lane] = mx;
      rrcp[lane] = 1.0f / sum;
    }
    __syncthreads();
#pragma unroll
    for (int r = 0; r < 8; ++r) {
      const int m = r + 8 * hi;
      float p = __expf(sc[m][ln] - rmax[m]) * rrcp[m];
      attnp[m][ln]      = (_Float16)p;
      attnp[m][16 + ln] = (_Float16)0.0f;   // K-pad 16..31 so one wmma covers K=16
    }
    __syncthreads();
#pragma unroll 1
    for (int dt = 0; dt < 2; ++dt) {
      half16_t A2 = load_a_frag(&attnp[ln][0], hi);            // attn 16x32 (zero-padded)
      half16_t B2 = {};
      if (hi == 0)                                             // rows K=0..15 = tokens; K>=16 x 0
        B2 = *(const half16_t*)&Vt[h * DH + dt * 16 + ln][0];
      float8_t c2 = {};
      c2 = __builtin_amdgcn_wmma_f32_16x16x32_f16(false, A2, false, B2,
                                                  (short)0, c2, false, false);
#pragma unroll
      for (int r = 0; r < 8; ++r)
        Xs[r + 8 * hi][h * DH + dt * 16 + ln] = (_Float16)c2[r];   // ctx reuses Xs
    }
    __syncthreads();
  }

  // ---- O-projection (ctx @ W0) + scatter fp32 results ----
#pragma unroll
  for (int s = 0; s < 8; ++s) Af[s] = load_a_frag(&Xs[ln][s * 32], hi);  // ctx fragments

#pragma unroll 1
  for (int t = 0; t < 16; ++t) {
    float8_t acc = {};
#pragma unroll
    for (int s = 0; s < 8; ++s) {
      const _Float16* bp = pw + ((((size_t)3 * 16 + t) * 8 + s) * 32 + lane) * 16;
      half16_t Bv = *(const half16_t*)bp;
      acc = __builtin_amdgcn_wmma_f32_16x16x32_f16(false, Af[s], false, Bv,
                                                   (short)0, acc, false, false);
    }
#pragma unroll
    for (int r = 0; r < 8; ++r) {
      const int m = r + 8 * hi;
      const int tok = sidx[m];
      if (tok >= 0)
        out[((size_t)(b * SS + tok)) * EE + t * 16 + ln] = acc[r];
    }
  }
}

// ---------------------------------------------------------------------------
// Launch
// d_in order: embeddings, Wq, Wk, Wv, W0, positions, num_positions, group_size, num_heads
// Workspace: [0, 512KB) packed f16 weights, [512KB, 1MB) group token indices.
// ---------------------------------------------------------------------------
extern "C" void kernel_launch(void* const* d_in, const int* in_sizes, int n_in,
                              void* d_out, int out_size, void* d_ws, size_t ws_size,
                              hipStream_t stream) {
  const float* emb = (const float*)d_in[0];
  const float* wq  = (const float*)d_in[1];
  const float* wk  = (const float*)d_in[2];
  const float* wv  = (const float*)d_in[3];
  const float* w0  = (const float*)d_in[4];
  const int* positions = (const int*)d_in[5];

  _Float16* packedW = (_Float16*)d_ws;                               // 4*65536 halfs = 512KB
  int* src_idx = (int*)((char*)d_ws + (size_t)4 * 65536 * sizeof(_Float16)); // 512KB

  pack_weights<<<(4 * 65536 + 255) / 256, 256, 0, stream>>>(wq, wk, wv, w0, packedW);
  build_groups<<<BB, PP, 0, stream>>>(positions, src_idx);
  pw_attn<<<GG, 32, 0, stream>>>(emb, packedW, src_idx, (float*)d_out);
}
